// MotionDecoder_81604378624348
// MI455X (gfx1250) — matrix operand order repaired and checked by reading, not compiled, not run on hardware
//
#include <hip/hip_runtime.h>
#include <cstdint>
#include <cstddef>

// ---------------------------------------------------------------------------
// Types for CDNA5 WMMA (wave32)
// ---------------------------------------------------------------------------
typedef __bf16 bf16;
typedef __attribute__((ext_vector_type(16))) __bf16 v16bf;
typedef __attribute__((ext_vector_type(8)))  float  v8f;
typedef __attribute__((ext_vector_type(4)))  float  f32x4;

#define WMMA_BF16(a, b, c) \
  __builtin_amdgcn_wmma_f32_16x16x32_bf16(false, (a), false, (b), (short)0, (c), false, false)

// Problem constants (fixed by the reference harness)
constexpr int Bc = 32, Sc = 128, Ec = 512, Hc = 512, NHc = 8, DHc = 64;
constexpr int Rc = 4, Tc = 512, Lc = 4, Jc = 50, OUTc = 2 * Jc + 2;  // 102
constexpr long BTc = (long)Bc * Tc;                                   // 16384

// ---------------------------------------------------------------------------
// Fragment loaders (16-bit A 16x32 / B 32x16; this file's contiguous k order, same for A and B)
// A: lane&15 = row M, halves = K(hi*16 .. +15) contiguous along ld
// B: lane&15 = column N, halves = K(hi*16 .. +15) contiguous along ld
// ---------------------------------------------------------------------------
__device__ inline v16bf load_a_frag(const bf16* base, int ld, int lane) {
  int r = lane & 15, hi = lane >> 4;
  const bf16* p = base + (size_t)r * ld + hi * 16;
  union { v16bf v; f32x4 q[2]; } u;
  u.q[0] = *(const f32x4*)(p);
  u.q[1] = *(const f32x4*)(p + 8);
  return u.v;
}

__device__ inline v16bf load_b_frag(const bf16* base, int ld, int lane) {
  int n = lane & 15, hi = lane >> 4;
  const bf16* p = base + (size_t)n * ld + hi * 16;
  union { v16bf v; f32x4 q[2]; } u;
  u.q[0] = *(const f32x4*)(p);
  u.q[1] = *(const f32x4*)(p + 8);
  return u.v;
}

__device__ inline float gelu_exact(float x) {
  return 0.5f * x * (1.0f + erff(x * 0.70710678118654752f));
}

__device__ inline float pe_val(int t, int d) {
  int i = d >> 1;
  float f = __expf((float)(2 * i) * (-9.210340371976184f / (float)Hc)); // ln(10000)
  float a = (float)t * f;
  return (d & 1) ? __cosf(a) : __sinf(a);
}

// ---------------------------------------------------------------------------
// Generic bf16 WMMA GEMM:  out = act(A @ W^T + bias [+ residual])
//   A:  M x K  bf16, row-major, lda          (M % 128 == 0 for all launches)
//   W:  N x K  bf16, row-major, ldw          (computes A @ W^T; N guarded)
//   block = 256 thr = 8 waves, block tile 128x64, wave tile 32x32 (2x2 WMMA)
//   The 64x32 weight tile is double-buffered in LDS via the gfx1250 async
//   global->LDS DMA path (ASYNCcnt): one barrier per K-step, DMA of tile i+1
//   overlaps the WMMAs of tile i.
// ---------------------------------------------------------------------------
__global__ __launch_bounds__(256)
void gemm_bf16_kernel(const bf16* __restrict__ A, int lda,
                      const bf16* __restrict__ W, int ldw,
                      const float* __restrict__ bias,
                      const float* __restrict__ residual, int ldr,
                      float* __restrict__ outF, int ldo,
                      bf16* __restrict__ outB, int ldb,
                      int M, int N, int K, int act) {
  __shared__ bf16 bTile[2][64 * 32];       // 2 x 4 KB weight tiles
  const int tid  = threadIdx.x;
  const int lane = tid & 31;
  const int wid  = tid >> 5;
  const int wm = wid >> 1, wn = wid & 1;
  const int m0   = blockIdx.y * 128 + wm * 32;
  const int nBlk = blockIdx.x * 64;

  // Async-stage assignment: each thread owns one 16 B chunk of the B tile.
  const int srow = tid >> 2;               // 0..63  (weight row within tile)
  const int schk = (tid & 3) * 8;          // half-offset 0,8,16,24
  const bool srowOK = (nBlk + srow) < N;
  const bf16* wRow = W + (size_t)(nBlk + srow) * ldw + schk;
  const uint32_t lds0 = (uint32_t)(uintptr_t)&bTile[0][srow * 32 + schk];
  const uint32_t lds1 = (uint32_t)(uintptr_t)&bTile[1][srow * 32 + schk];

  // Out-of-range weight rows: zero once; the DMA never touches them.
  if (!srowOK) {
    f32x4 z = {0.f, 0.f, 0.f, 0.f};
    *(f32x4*)&bTile[0][srow * 32 + schk] = z;
    *(f32x4*)&bTile[1][srow * 32 + schk] = z;
  }

  // Kick off DMA of the first tile.
  if (srowOK) {
    uint64_t ga = (uint64_t)(uintptr_t)(wRow);
    asm volatile("global_load_async_to_lds_b128 %0, %1, off"
                 :: "v"(lds0), "v"(ga) : "memory");
  }

  v8f acc[2][2] = {};
  const int nk = K / 32;
  for (int i = 0; i < nk; ++i) {
    const int k0 = i * 32;
    const int cur = i & 1;
    // A fragments straight from global while the DMA is in flight
    v16bf a0 = load_a_frag(A + (size_t)m0 * lda + k0, lda, lane);
    v16bf a1 = load_a_frag(A + (size_t)(m0 + 16) * lda + k0, lda, lane);
    asm volatile("s_wait_asynccnt 0x0" ::: "memory");
    __syncthreads();   // DMA_i visible everywhere; prior reads of both bufs done
    // B fragments from LDS (ds_load_b128)
    v16bf b0 = load_b_frag(&bTile[cur][(wn * 32 + 0)  * 32], 32, lane);
    v16bf b1 = load_b_frag(&bTile[cur][(wn * 32 + 16) * 32], 32, lane);
    // Launch DMA of the next tile into the other buffer (overlaps the WMMAs)
    if (i + 1 < nk && srowOK) {
      uint64_t ga = (uint64_t)(uintptr_t)(wRow + k0 + 32);
      asm volatile("global_load_async_to_lds_b128 %0, %1, off"
                   :: "v"(cur ? lds0 : lds1), "v"(ga) : "memory");
    }
    acc[0][0] = WMMA_BF16(a0, b0, acc[0][0]);
    acc[0][1] = WMMA_BF16(a0, b1, acc[0][1]);
    acc[1][0] = WMMA_BF16(a1, b0, acc[1][0]);
    acc[1][1] = WMMA_BF16(a1, b1, acc[1][1]);
  }

  const int r = lane & 15, hi = lane >> 4;
  const int n0 = nBlk + wn * 32;
#pragma unroll
  for (int mt = 0; mt < 2; ++mt)
#pragma unroll
    for (int nt = 0; nt < 2; ++nt) {
      int col = n0 + nt * 16 + r;
      if (col >= N) continue;
      float bv = bias ? bias[col] : 0.f;
#pragma unroll
      for (int e = 0; e < 8; ++e) {
        int row = m0 + mt * 16 + hi * 8 + e;
        float v = acc[mt][nt][e] + bv;
        if (residual) v += residual[(size_t)row * ldr + col];
        if (act) v = gelu_exact(v);
        if (outF) outF[(size_t)row * ldo + col] = v;
        if (outB) outB[(size_t)row * ldb + col] = (bf16)v;
      }
    }
}

// ---------------------------------------------------------------------------
// Fused flash attention: one wave handles a 16-query tile for one (b, head).
// S = Q K^T via WMMA, online softmax, P staged through LDS, P V via WMMA.
// ---------------------------------------------------------------------------
__global__ __launch_bounds__(32)
void attn_kernel(const bf16* __restrict__ Q, int q_rs,
                 const bf16* __restrict__ K, int k_rs,
                 const bf16* __restrict__ V, int v_rs,
                 bf16* __restrict__ O, int o_rs,
                 int Tq, int Tk, float scale) {
  __shared__ bf16 pbuf[16 * 32];
  const int lane = threadIdx.x;
  const int qt = blockIdx.x, h = blockIdx.y, b = blockIdx.z;
  const int hoff = h * DHc;
  const bf16* Qb = Q + ((size_t)b * Tq + qt * 16) * q_rs + hoff;
  const bf16* Kb = K + (size_t)b * Tk * k_rs + hoff;
  const bf16* Vb = V + (size_t)b * Tk * v_rs + hoff;
  const int r = lane & 15, hi = lane >> 4;

  // Q tile (16 x 64) as two A fragments (K chunks 0..31, 32..63)
  v16bf qa0 = load_a_frag(Qb,      q_rs, lane);
  v16bf qa1 = load_a_frag(Qb + 32, q_rs, lane);

  v8f o[4] = {};
  float mrow[8], lrow[8];
#pragma unroll
  for (int e = 0; e < 8; ++e) { mrow[e] = -1e30f; lrow[e] = 0.f; }

  for (int kk = 0; kk < Tk; kk += 32) {
    v8f s0 = {}, s1 = {};
    {
      v16bf kb;
      kb = load_b_frag(Kb + (size_t)kk * k_rs,             k_rs, lane);
      s0 = WMMA_BF16(qa0, kb, s0);
      kb = load_b_frag(Kb + (size_t)kk * k_rs + 32,        k_rs, lane);
      s0 = WMMA_BF16(qa1, kb, s0);
      kb = load_b_frag(Kb + (size_t)(kk + 16) * k_rs,      k_rs, lane);
      s1 = WMMA_BF16(qa0, kb, s1);
      kb = load_b_frag(Kb + (size_t)(kk + 16) * k_rs + 32, k_rs, lane);
      s1 = WMMA_BF16(qa1, kb, s1);
    }
    // Online softmax per C-layout row (row = hi*8+e lives in 16 lanes of half hi)
#pragma unroll
    for (int e = 0; e < 8; ++e) {
      float c0 = s0[e] * scale, c1 = s1[e] * scale;
      float mx = fmaxf(c0, c1);
      for (int d = 8; d >= 1; d >>= 1) mx = fmaxf(mx, __shfl_xor(mx, d, 32));
      float mnew = fmaxf(mrow[e], mx);
      float alpha = __expf(mrow[e] - mnew);
      float p0 = __expf(c0 - mnew), p1 = __expf(c1 - mnew);
      float sum = p0 + p1;
      for (int d = 8; d >= 1; d >>= 1) sum += __shfl_xor(sum, d, 32);
      lrow[e] = lrow[e] * alpha + sum;
      mrow[e] = mnew;
#pragma unroll
      for (int t = 0; t < 4; ++t) o[t][e] *= alpha;
      pbuf[(hi * 8 + e) * 32 + r]      = (bf16)p0;
      pbuf[(hi * 8 + e) * 32 + 16 + r] = (bf16)p1;
    }
    __syncthreads();
    v16bf pa = load_a_frag(pbuf, 32, lane);   // ds_load_b128 in A-frag layout
#pragma unroll
    for (int nt = 0; nt < 4; ++nt) {
      v16bf vb;
      const bf16* vp = Vb + (size_t)(kk + hi * 16) * v_rs + nt * 16 + r;
#pragma unroll
      for (int j = 0; j < 16; ++j) vb[j] = vp[(size_t)j * v_rs];
      o[nt] = WMMA_BF16(pa, vb, o[nt]);
    }
    __syncthreads();
  }

  bf16* Ob = O + ((size_t)b * Tq + qt * 16) * o_rs + hoff;
#pragma unroll
  for (int nt = 0; nt < 4; ++nt)
#pragma unroll
    for (int e = 0; e < 8; ++e) {
      float v = o[nt][e] / lrow[e];
      Ob[(size_t)(hi * 8 + e) * o_rs + nt * 16 + r] = (bf16)v;
    }
}

// ---------------------------------------------------------------------------
// LayerNorm over H=512, one row per 256-thread block, bf16 output
// ---------------------------------------------------------------------------
__global__ __launch_bounds__(256)
void ln_kernel(const float* __restrict__ X, const float* __restrict__ G,
               const float* __restrict__ Bb, bf16* __restrict__ Y) {
  const int row = blockIdx.x;
  const float* x = X + (size_t)row * Hc;
  float a = x[threadIdx.x], c = x[threadIdx.x + 256];
  float s = a + c, s2 = a * a + c * c;
  for (int d = 16; d >= 1; d >>= 1) {
    s  += __shfl_xor(s, d, 32);
    s2 += __shfl_xor(s2, d, 32);
  }
  __shared__ float sh[16];
  int wid = threadIdx.x >> 5, lane = threadIdx.x & 31;
  if (lane == 0) { sh[wid] = s; sh[8 + wid] = s2; }
  __syncthreads();
  if (wid == 0) {
    float ts  = (lane < 8) ? sh[lane] : 0.f;
    float ts2 = (lane < 8) ? sh[8 + lane] : 0.f;
    for (int d = 4; d >= 1; d >>= 1) {
      ts  += __shfl_xor(ts, d, 32);
      ts2 += __shfl_xor(ts2, d, 32);
    }
    if (lane == 0) { sh[0] = ts; sh[1] = ts2; }
  }
  __syncthreads();
  float mean = sh[0] / Hc;
  float var  = sh[1] / Hc - mean * mean;
  float rstd = rsqrtf(var + 1e-5f);
  bf16* y = Y + (size_t)row * Hc;
  int i0 = threadIdx.x, i1 = threadIdx.x + 256;
  y[i0] = (bf16)((a - mean) * rstd * G[i0] + Bb[i0]);
  y[i1] = (bf16)((c - mean) * rstd * G[i1] + Bb[i1]);
}

// ---------------------------------------------------------------------------
// Elementwise helpers
// ---------------------------------------------------------------------------
__global__ void cvt_bf16_kernel(const float* __restrict__ in, bf16* __restrict__ out, long n) {
  long i = (long)blockIdx.x * blockDim.x + threadIdx.x;
  if (i < n) out[i] = (bf16)in[i];
}

// wconv[k][d][c] = conv_w[c][d][k]  (conv_w is (C=512, D=512, R=4) f32)
__global__ void convw_kernel(const float* __restrict__ cw, bf16* __restrict__ out) {
  long i = (long)blockIdx.x * blockDim.x + threadIdx.x;
  long total = (long)Rc * Hc * Hc;
  if (i >= total) return;
  int c = (int)(i % Hc);
  long t = i / Hc;
  int d = (int)(t % Hc);
  int k = (int)(t / Hc);
  out[i] = (bf16)cw[((size_t)c * Hc + d) * Rc + k];
}

// mem = conv_out + conv_b + PE  -> bf16
__global__ void memfin_kernel(const float* __restrict__ co, const float* __restrict__ cb,
                              bf16* __restrict__ mem) {
  long i = (long)blockIdx.x * blockDim.x + threadIdx.x;
  if (i >= BTc * Hc) return;
  int d = (int)(i % Hc);
  int t = (int)((i / Hc) % Tc);
  mem[i] = (bf16)(co[i] + cb[d] + pe_val(t, d));
}

// x = query_emb (broadcast over B) + PE  -> f32
__global__ void xinit_kernel(const float* __restrict__ qe, float* __restrict__ x) {
  long i = (long)blockIdx.x * blockDim.x + threadIdx.x;
  if (i >= BTc * Hc) return;
  int d = (int)(i % Hc);
  int t = (int)((i / Hc) % Tc);
  x[i] = qe[(size_t)t * Hc + d] + pe_val(t, d);
}

// ---------------------------------------------------------------------------
// Forward kinematics: one thread per (b, t)
// ---------------------------------------------------------------------------
__constant__ int PAR[Jc] = {
  -1, 0, 1, 2, 3, 1, 5, 6, 1,
   4, 9, 10, 11,  4, 13, 14, 15,  4, 17, 18, 19,  4, 21, 22, 23,  4, 25, 26, 27,
   7, 29, 30, 31, 7, 33, 34, 35,  7, 37, 38, 39,  7, 41, 42, 43,
   8};

__global__ void kin_kernel(const float* __restrict__ outp, float* __restrict__ pose) {
  long i = (long)blockIdx.x * blockDim.x + threadIdx.x;
  if (i >= BTc) return;
  const float* row = outp + i * OUTc;
  float g[Jc], px[Jc], py[Jc];
  g[0] = 0.f; px[0] = row[2 * Jc]; py[0] = row[2 * Jc + 1];
  for (int j = 1; j < Jc; ++j) {
    int p = PAR[j];
    float gg = g[p] + row[j];
    g[j] = gg;
    float s = row[Jc + j];
    float sn, cs;
    __sincosf(gg, &sn, &cs);
    px[j] = px[p] + cs * s;
    py[j] = py[p] + sn * s;
  }
  float* o = pose + i * (3 * Jc);
  for (int j = 0; j < Jc; ++j) {
    o[3 * j + 0] = px[j];
    o[3 * j + 1] = py[j];
    o[3 * j + 2] = 1.f;
  }
}

// ---------------------------------------------------------------------------
// Host orchestration
// ---------------------------------------------------------------------------
extern "C" void kernel_launch(void* const* d_in, const int* in_sizes, int n_in,
                              void* d_out, int out_size, void* d_ws, size_t ws_size,
                              hipStream_t stream) {
  (void)in_sizes; (void)n_in; (void)out_size; (void)ws_size;
  const float* memory    = (const float*)d_in[0];
  const float* in_proj_w = (const float*)d_in[1];
  const float* in_proj_b = (const float*)d_in[2];
  const float* conv_w    = (const float*)d_in[3];
  const float* conv_b    = (const float*)d_in[4];
  const float* query_emb = (const float*)d_in[5];
  const float* sa_in_w   = (const float*)d_in[6];
  const float* sa_in_b   = (const float*)d_in[7];
  const float* sa_out_w  = (const float*)d_in[8];
  const float* sa_out_b  = (const float*)d_in[9];
  const float* ca_in_w   = (const float*)d_in[10];
  const float* ca_in_b   = (const float*)d_in[11];
  const float* ca_out_w  = (const float*)d_in[12];
  const float* ca_out_b  = (const float*)d_in[13];
  const float* ln1_g = (const float*)d_in[14];
  const float* ln1_b = (const float*)d_in[15];
  const float* ln2_g = (const float*)d_in[16];
  const float* ln2_b = (const float*)d_in[17];
  const float* ln3_g = (const float*)d_in[18];
  const float* ln3_b = (const float*)d_in[19];
  const float* ff1_w = (const float*)d_in[20];
  const float* ff1_b = (const float*)d_in[21];
  const float* ff2_w = (const float*)d_in[22];
  const float* ff2_b = (const float*)d_in[23];
  const float* out_w = (const float*)d_in[24];
  const float* out_b = (const float*)d_in[25];

  // Bump allocator over d_ws
  char* cur = (char*)d_ws;
  auto alloc = [&](size_t bytes) -> void* {
    void* p = (void*)cur;
    cur += (bytes + 255) & ~(size_t)255;
    return p;
  };

  const long BT = BTc;
  bf16* memB   = (bf16*)alloc((size_t)Bc * Sc * Ec * 2);
  bf16* wInP   = (bf16*)alloc((size_t)Hc * Ec * 2);
  bf16* wConv  = (bf16*)alloc((size_t)Rc * Hc * Hc * 2);
  bf16* wSaIn  = (bf16*)alloc((size_t)Lc * 3 * Hc * Hc * 2);
  bf16* wSaOut = (bf16*)alloc((size_t)Lc * Hc * Hc * 2);
  bf16* wCaIn  = (bf16*)alloc((size_t)Lc * 3 * Hc * Hc * 2);
  bf16* wCaOut = (bf16*)alloc((size_t)Lc * Hc * Hc * 2);
  bf16* wFf1   = (bf16*)alloc((size_t)Lc * 4 * Hc * Hc * 2);
  bf16* wFf2   = (bf16*)alloc((size_t)Lc * Hc * 4 * Hc * 2);
  bf16* wOut   = (bf16*)alloc((size_t)OUTc * Hc * 2);
  bf16* mB     = (bf16*)alloc((size_t)Bc * Sc * Hc * 2);
  float* convO = (float*)alloc((size_t)BT * Hc * 4);
  bf16* memKV  = (bf16*)alloc((size_t)BT * Hc * 2);
  float* x     = (float*)alloc((size_t)BT * Hc * 4);
  bf16* hB     = (bf16*)alloc((size_t)BT * Hc * 2);
  bf16* qkv    = (bf16*)alloc((size_t)BT * 3 * Hc * 2);
  bf16* attnB  = (bf16*)alloc((size_t)BT * Hc * 2);
  bf16* f1     = (bf16*)alloc((size_t)BT * 4 * Hc * 2);
  float* outP  = (float*)alloc((size_t)BT * OUTc * 4);

  auto cvt = [&](const float* in, bf16* out, long n) {
    cvt_bf16_kernel<<<dim3((unsigned)((n + 255) / 256)), dim3(256), 0, stream>>>(in, out, n);
  };
  auto gemm = [&](const bf16* A, int lda, const bf16* W, int ldw, const float* bias,
                  const float* res, int ldr, float* oF, int ldo, bf16* oB, int ldb,
                  int M, int N, int K, int act) {
    dim3 g((N + 63) / 64, M / 128), blk(256);
    gemm_bf16_kernel<<<g, blk, 0, stream>>>(A, lda, W, ldw, bias, res, ldr,
                                            oF, ldo, oB, ldb, M, N, K, act);
  };

  // Weight / input conversion to bf16
  cvt(memory,    memB,   (long)Bc * Sc * Ec);
  cvt(in_proj_w, wInP,   (long)Hc * Ec);
  convw_kernel<<<dim3((unsigned)(((long)Rc * Hc * Hc + 255) / 256)), dim3(256), 0, stream>>>(conv_w, wConv);
  cvt(sa_in_w,  wSaIn,  (long)Lc * 3 * Hc * Hc);
  cvt(sa_out_w, wSaOut, (long)Lc * Hc * Hc);
  cvt(ca_in_w,  wCaIn,  (long)Lc * 3 * Hc * Hc);
  cvt(ca_out_w, wCaOut, (long)Lc * Hc * Hc);
  cvt(ff1_w,    wFf1,   (long)Lc * 4 * Hc * Hc);
  cvt(ff2_w,    wFf2,   (long)Lc * Hc * 4 * Hc);
  cvt(out_w,    wOut,   (long)OUTc * Hc);

  // m = memory @ in_proj_w^T + b   (bf16 out)
  gemm(memB, Ec, wInP, Ec, in_proj_b, nullptr, 0, nullptr, 0, mB, Hc,
       Bc * Sc, Hc, Ec, 0);
  // conv upsample as 4 scattered GEMMs: out row t = 4s + k
  for (int k = 0; k < Rc; ++k)
    gemm(mB, Hc, wConv + (size_t)k * Hc * Hc, Hc, nullptr, nullptr, 0,
         convO + (size_t)k * Hc, Rc * Hc, nullptr, 0, Bc * Sc, Hc, Hc, 0);
  memfin_kernel<<<dim3((unsigned)((BT * Hc + 255) / 256)), dim3(256), 0, stream>>>(convO, conv_b, memKV);
  xinit_kernel<<<dim3((unsigned)((BT * Hc + 255) / 256)), dim3(256), 0, stream>>>(query_emb, x);

  const float scale = 0.125f;  // 1/sqrt(dh=64)
  bf16* qCA  = qkv;
  bf16* kvCA = qkv + (size_t)BT * Hc;

  for (int l = 0; l < Lc; ++l) {
    // --- self attention ---
    ln_kernel<<<dim3((unsigned)BT), dim3(256), 0, stream>>>(x, ln1_g + l * Hc, ln1_b + l * Hc, hB);
    gemm(hB, Hc, wSaIn + (size_t)l * 3 * Hc * Hc, Hc, sa_in_b + l * 3 * Hc,
         nullptr, 0, nullptr, 0, qkv, 3 * Hc, (int)BT, 3 * Hc, Hc, 0);
    attn_kernel<<<dim3(Tc / 16, NHc, Bc), dim3(32), 0, stream>>>(
        qkv, 3 * Hc, qkv + Hc, 3 * Hc, qkv + 2 * Hc, 3 * Hc,
        attnB, Hc, Tc, Tc, scale);
    gemm(attnB, Hc, wSaOut + (size_t)l * Hc * Hc, Hc, sa_out_b + l * Hc,
         x, Hc, x, Hc, nullptr, 0, (int)BT, Hc, Hc, 0);

    // --- cross attention ---
    ln_kernel<<<dim3((unsigned)BT), dim3(256), 0, stream>>>(x, ln2_g + l * Hc, ln2_b + l * Hc, hB);
    gemm(hB, Hc, wCaIn + (size_t)l * 3 * Hc * Hc, Hc, ca_in_b + l * 3 * Hc,
         nullptr, 0, nullptr, 0, qCA, Hc, (int)BT, Hc, Hc, 0);
    gemm(memKV, Hc, wCaIn + (size_t)l * 3 * Hc * Hc + (size_t)Hc * Hc, Hc,
         ca_in_b + l * 3 * Hc + Hc, nullptr, 0, nullptr, 0, kvCA, 2 * Hc,
         (int)BT, 2 * Hc, Hc, 0);
    attn_kernel<<<dim3(Tc / 16, NHc, Bc), dim3(32), 0, stream>>>(
        qCA, Hc, kvCA, 2 * Hc, kvCA + Hc, 2 * Hc, attnB, Hc, Tc, Tc, scale);
    gemm(attnB, Hc, wCaOut + (size_t)l * Hc * Hc, Hc, ca_out_b + l * Hc,
         x, Hc, x, Hc, nullptr, 0, (int)BT, Hc, Hc, 0);

    // --- FFN ---
    ln_kernel<<<dim3((unsigned)BT), dim3(256), 0, stream>>>(x, ln3_g + l * Hc, ln3_b + l * Hc, hB);
    gemm(hB, Hc, wFf1 + (size_t)l * 4 * Hc * Hc, Hc, ff1_b + l * 4 * Hc,
         nullptr, 0, nullptr, 0, f1, 4 * Hc, (int)BT, 4 * Hc, Hc, 1 /*gelu*/);
    gemm(f1, 4 * Hc, wFf2 + (size_t)l * Hc * 4 * Hc, 4 * Hc, ff2_b + l * Hc,
         x, Hc, x, Hc, nullptr, 0, (int)BT, Hc, 4 * Hc, 0);
  }

  // output projection (N=102, guarded) + forward kinematics
  cvt(x, hB, BT * Hc);
  gemm(hB, Hc, wOut, Hc, out_b, nullptr, 0, outP, OUTc, nullptr, 0,
       (int)BT, OUTc, Hc, 0);
  kin_kernel<<<dim3((unsigned)((BT + 255) / 256)), dim3(256), 0, stream>>>(outP, (float*)d_out);
}
